// CountingAbstraction_39058432589956
// MI455X (gfx1250) — compile-verified, matrix operand order checked by reading, not run on hardware
//
#include <hip/hip_runtime.h>
#include <hip/hip_bf16.h>

typedef __attribute__((ext_vector_type(16))) _Float16 v16h;
typedef __attribute__((ext_vector_type(8)))  _Float16 v8h;
typedef __attribute__((ext_vector_type(8)))  float    v8f;
typedef __attribute__((ext_vector_type(4)))  int      v4i;

#define BM 128
#define BN 128
#define BK 32
#define LDT 40          // padded LDS row stride (halfs): 80B, 16B aligned, conflict-free
#define THREADS 256     // helper kernels
#define GT 128          // gemm kernel: 4 waves, each owning a 64x64 C tile (4x4 WMMA frags)

#define AS1 __attribute__((address_space(1)))
#define AS3 __attribute__((address_space(3)))

#if defined(__AMDGCN__) && __has_builtin(__builtin_amdgcn_global_load_async_to_lds_b128)
#define HAVE_ASYNC_LDS 1
#else
#define HAVE_ASYNC_LDS 0
#endif

static __device__ __forceinline__ void wait_async_lds() {
#if defined(__AMDGCN__)
#if __has_builtin(__builtin_amdgcn_s_wait_asynccnt)
  __builtin_amdgcn_s_wait_asynccnt(0);
#else
  asm volatile("s_wait_asynccnt 0x0" ::: "memory");
#endif
#endif
}

// ---------------- l2-normalize rows + f16 conversion ----------------
__global__ __launch_bounds__(THREADS)
void k_norm(const float* __restrict__ data, _Float16* __restrict__ cn,
            _Float16* __restrict__ acat, int D, int acatLd) {
  __shared__ float red[THREADS];
  const size_t row = blockIdx.x;
  const float* src = data + row * (size_t)D;
  float s = 0.f;
  for (int i = threadIdx.x; i < D; i += THREADS) { float v = src[i]; s += v * v; }
  red[threadIdx.x] = s; __syncthreads();
  for (int off = THREADS / 2; off > 0; off >>= 1) {
    if (threadIdx.x < off) red[threadIdx.x] += red[threadIdx.x + off];
    __syncthreads();
  }
  const float scale = rsqrtf(fmaxf(red[0], 1e-12f));
  _Float16* cdst = cn + row * (size_t)D;
  _Float16* adst = acat + row * (size_t)acatLd;
  for (int i = threadIdx.x; i < D; i += THREADS) {
    float v = src[i];
    cdst[i] = (_Float16)(v * scale);
    adst[i] = (_Float16)v;
  }
}

// ---------------- sinusoidal positional encoding (f16) ----------------
__global__ __launch_bounds__(THREADS)
void k_posenc(_Float16* __restrict__ pe, int N, int D) {
  const int n = blockIdx.x;
  const float c = 9.210340371976184f / (float)D;  // ln(10000)/D
  for (int d = threadIdx.x; d < D; d += THREADS) {
    int i2 = d & ~1;
    float freq = __expf(-(float)i2 * c);
    float ang  = (float)n * freq;
    float v    = (d & 1) ? __cosf(ang) : __sinf(ang);
    pe[(size_t)n * D + d] = (_Float16)v;
  }
}

// ---------------- row sums of sim (f16 -> f32) ----------------
__global__ __launch_bounds__(THREADS)
void k_rowsum(const _Float16* __restrict__ sim, float* __restrict__ csum, int N) {
  __shared__ float red[THREADS];
  const size_t row = blockIdx.x;
  const _Float16* src = sim + row * (size_t)N;
  float s = 0.f;
  for (int i = threadIdx.x; i < N; i += THREADS) s += (float)src[i];
  red[threadIdx.x] = s; __syncthreads();
  for (int off = THREADS / 2; off > 0; off >>= 1) {
    if (threadIdx.x < off) red[threadIdx.x] += red[threadIdx.x + off];
    __syncthreads();
  }
  if (threadIdx.x == 0) csum[row] = red[0];
}

// ---------------- tiled f32 -> f16 transpose: out[c][r] = in[r+rowOff][c] ----------------
__global__ void k_transpose(const float* __restrict__ in, int rows, int cols, int rowOff,
                            _Float16* __restrict__ out) {
  __shared__ float tile[32][33];
  const int c = blockIdx.x * 32 + threadIdx.x;
  #pragma unroll
  for (int j = 0; j < 4; ++j) {
    int r = blockIdx.y * 32 + threadIdx.y + j * 8;
    if (r < rows && c < cols) tile[threadIdx.y + j * 8][threadIdx.x] = in[(size_t)(r + rowOff) * cols + c];
  }
  __syncthreads();
  const int r2 = blockIdx.y * 32 + threadIdx.x;
  #pragma unroll
  for (int j = 0; j < 4; ++j) {
    int c2 = blockIdx.x * 32 + threadIdx.y + j * 8;
    if (r2 < rows && c2 < cols) out[(size_t)c2 * rows + r2] = (_Float16)tile[threadIdx.x][threadIdx.y + j * 8];
  }
}

__global__ void k_copy_row0(const float* __restrict__ W, float* __restrict__ w0, int D) {
  int i = blockIdx.x * THREADS + threadIdx.x;
  if (i < D) w0[i] = W[i];
}

// ---------------- WMMA GEMM: C = epilogue( A[M,K] @ BT[N,K]^T ) ----------------
// 128x128 block, 4 waves, 64x64 per wave (4x4 16x16x32 WMMA frags) -> 1 ds_load_b128 per WMMA.
// MODE 0: relu -> f16        MODE 1: raw -> f16
// MODE 2: +csum[m]*w0[n]+bexp[n], softplus -> f16    MODE 3: raw -> f32
template <int MODE>
__global__ __launch_bounds__(GT)
void k_gemm(const _Float16* __restrict__ A, long long aStride, int lda,
            const _Float16* __restrict__ BT, long long bStride, int ldb,
            void* __restrict__ C, long long cStride, int ldc, int K,
            const float* __restrict__ csum, const float* __restrict__ w0,
            const float* __restrict__ bexp) {
  __shared__ _Float16 lsA[2][BM * LDT];
  __shared__ _Float16 lsB[2][BN * LDT];

  const int z = blockIdx.z;
  A  += (size_t)z * aStride;
  BT += (size_t)z * bStride;

  const int t = threadIdx.x;

  // Staging map: 128 rows x 4 chunks (8 halfs) per matrix, 4 chunks per thread,
  // groups of 4 consecutive lanes cover one 64B row segment (coalesced).
  int srow[4], scol[4];
  size_t aOff[4], bOff[4];
  #pragma unroll
  for (int i = 0; i < 4; ++i) {
    int ch = t + GT * i;
    srow[i] = ch >> 2;
    scol[i] = (ch & 3) * 8;
    aOff[i] = (size_t)(blockIdx.y * BM + srow[i]) * lda + scol[i];
    bOff[i] = (size_t)(blockIdx.x * BN + srow[i]) * ldb + scol[i];
  }

#if !HAVE_ASYNC_LDS
  v8h ra[4], rb[4];
#endif

  // prologue: k-tile 0 -> LDS buffer 0
#if HAVE_ASYNC_LDS
  #pragma unroll
  for (int i = 0; i < 4; ++i) {
    __builtin_amdgcn_global_load_async_to_lds_b128(
        (AS1 v4i*)(A + aOff[i]), (AS3 v4i*)&lsA[0][srow[i] * LDT + scol[i]], 0, 0);
    __builtin_amdgcn_global_load_async_to_lds_b128(
        (AS1 v4i*)(BT + bOff[i]), (AS3 v4i*)&lsB[0][srow[i] * LDT + scol[i]], 0, 0);
  }
  wait_async_lds();
#else
  #pragma unroll
  for (int i = 0; i < 4; ++i) {
    ra[i] = *(const v8h*)(A + aOff[i]);
    rb[i] = *(const v8h*)(BT + bOff[i]);
  }
  #pragma unroll
  for (int i = 0; i < 4; ++i) {
    *(v8h*)&lsA[0][srow[i] * LDT + scol[i]] = ra[i];
    *(v8h*)&lsB[0][srow[i] * LDT + scol[i]] = rb[i];
  }
#endif

  const int lane = t & 31;
  const int wave = t >> 5;
  const int wm   = wave >> 1;        // 0..1 (64-row slab)
  const int wn   = wave & 1;         // 0..1 (64-col slab)
  const int lr   = lane & 15;
  const int kh   = (lane >> 4) * 8;  // half-index offset per ISA A/B frag layout

  v8f zero = {};
  v8f acc[4][4];
  #pragma unroll
  for (int i = 0; i < 4; ++i)
    #pragma unroll
    for (int j = 0; j < 4; ++j) acc[i][j] = zero;

  const int KT = K / BK;
  for (int kt = 0; kt < KT; ++kt) {
    __syncthreads();
    const int cur = kt & 1;
    const int nxt = cur ^ 1;
    if (kt + 1 < KT) {  // stage next k-tile while this one computes
      const size_t ko = (size_t)(kt + 1) * BK;
#if HAVE_ASYNC_LDS
      #pragma unroll
      for (int i = 0; i < 4; ++i) {
        __builtin_amdgcn_global_load_async_to_lds_b128(
            (AS1 v4i*)(A + aOff[i] + ko), (AS3 v4i*)&lsA[nxt][srow[i] * LDT + scol[i]], 0, 0);
        __builtin_amdgcn_global_load_async_to_lds_b128(
            (AS1 v4i*)(BT + bOff[i] + ko), (AS3 v4i*)&lsB[nxt][srow[i] * LDT + scol[i]], 0, 0);
      }
#else
      #pragma unroll
      for (int i = 0; i < 4; ++i) {
        ra[i] = *(const v8h*)(A + aOff[i] + ko);
        rb[i] = *(const v8h*)(BT + bOff[i] + ko);
      }
#endif
    }

    v16h af[4], bf[4];
    #pragma unroll
    for (int fm = 0; fm < 4; ++fm) {
      const _Float16* p = &lsA[cur][(wm * 64 + fm * 16 + lr) * LDT + kh];
      v8h lo = *(const v8h*)p;
      v8h hi = *(const v8h*)(p + 16);
      af[fm] = __builtin_shufflevector(lo, hi, 0,1,2,3,4,5,6,7,8,9,10,11,12,13,14,15);
    }
    #pragma unroll
    for (int fn = 0; fn < 4; ++fn) {
      const _Float16* p = &lsB[cur][(wn * 64 + fn * 16 + lr) * LDT + kh];
      v8h lo = *(const v8h*)p;
      v8h hi = *(const v8h*)(p + 16);
      bf[fn] = __builtin_shufflevector(lo, hi, 0,1,2,3,4,5,6,7,8,9,10,11,12,13,14,15);
    }
    #pragma unroll
    for (int fm = 0; fm < 4; ++fm)
      #pragma unroll
      for (int fn = 0; fn < 4; ++fn)
        acc[fm][fn] = __builtin_amdgcn_wmma_f32_16x16x32_f16(
            false, af[fm], false, bf[fn], (short)0, acc[fm][fn], false, false);

    if (kt + 1 < KT) {
#if HAVE_ASYNC_LDS
      wait_async_lds();          // drain our async writes to buf[nxt] before barrier
#else
      #pragma unroll
      for (int i = 0; i < 4; ++i) {
        *(v8h*)&lsA[nxt][srow[i] * LDT + scol[i]] = ra[i];
        *(v8h*)&lsB[nxt][srow[i] * LDT + scol[i]] = rb[i];
      }
#endif
    }
  }

  // ---------------- epilogue ----------------
  const int gmBase = blockIdx.y * BM + wm * 64;
  const int gnBase = blockIdx.x * BN + wn * 64;
  const int eRow   = (lane >> 4) * 8;
  const int Mtot   = gridDim.y * BM;
  #pragma unroll
  for (int fm = 0; fm < 4; ++fm) {
    #pragma unroll
    for (int fn = 0; fn < 4; ++fn) {
      const int col = gnBase + fn * 16 + lr;
      #pragma unroll
      for (int e = 0; e < 8; ++e) {
        const int row = gmBase + fm * 16 + eRow + e;
        float v = acc[fm][fn][e];
        if (MODE == 0) {
          _Float16* Cp = (_Float16*)C + (size_t)z * cStride;
          Cp[(size_t)row * ldc + col] = (_Float16)fmaxf(v, 0.f);
        } else if (MODE == 1) {
          _Float16* Cp = (_Float16*)C + (size_t)z * cStride;
          Cp[(size_t)row * ldc + col] = (_Float16)v;
        } else if (MODE == 2) {
          v += csum[(size_t)z * Mtot + row] * w0[col] + bexp[col];
          v = (v > 20.f) ? v : log1pf(__expf(v));   // softplus
          _Float16* Cp = (_Float16*)C + (size_t)z * cStride;
          Cp[(size_t)row * ldc + col] = (_Float16)v;
        } else {
          float* Cp = (float*)C + (size_t)z * cStride;
          Cp[(size_t)row * ldc + col] = v;
        }
      }
    }
  }
}

extern "C" void kernel_launch(void* const* d_in, const int* in_sizes, int n_in,
                              void* d_out, int out_size, void* d_ws, size_t ws_size,
                              hipStream_t stream) {
  (void)in_sizes; (void)n_in; (void)out_size; (void)ws_size;
  constexpr int B = 4, N = 2048, D = 2048;
  const float* data    = (const float*)d_in[0];   // [B,N,D]
  const float* W_exp   = (const float*)d_in[1];   // [D+1,D]
  const float* b_exp   = (const float*)d_in[2];   // [D]
  const float* W_merge = (const float*)d_in[3];   // [2D,D]

  // workspace layout (256B-aligned chunks)
  char* ws = (char*)d_ws;
  size_t off = 0;
  auto take = [&](size_t bytes) { char* p = ws + off; off += (bytes + 255) & ~(size_t)255; return p; };
  _Float16* cn      = (_Float16*)take((size_t)B * N * D * 2);       // aliased as vout later
  _Float16* acat    = (_Float16*)take((size_t)B * N * 2 * D * 2);   // [data | counter] f16
  _Float16* sim     = (_Float16*)take((size_t)B * N * N * 2);
  _Float16* pe      = (_Float16*)take((size_t)N * D * 2);
  _Float16* WexpT   = (_Float16*)take((size_t)D * D * 2);           // [j][k] = W_exp[1+k][j]
  _Float16* WmergeT = (_Float16*)take((size_t)D * 2 * D * 2);       // [j][k] = W_merge[k][j]
  float*    w0      = (float*)take((size_t)D * 4);
  float*    csum    = (float*)take((size_t)B * N * 4);
  _Float16* vout    = cn;  // cn is dead after GEMM1; reuse for v_output

  const dim3 tb(THREADS);
  const dim3 tg(GT);
  const dim3 gGemm(N / BN, N / BM, B);

  // prep
  k_copy_row0<<<(D + THREADS - 1) / THREADS, tb, 0, stream>>>(W_exp, w0, D);
  k_transpose<<<dim3(D / 32, D / 32), dim3(32, 8), 0, stream>>>(W_exp, D, D, 1, WexpT);
  k_transpose<<<dim3(D / 32, (2 * D) / 32), dim3(32, 8), 0, stream>>>(W_merge, 2 * D, D, 0, WmergeT);
  k_norm<<<B * N, tb, 0, stream>>>(data, cn, acat, D, 2 * D);
  k_posenc<<<N, tb, 0, stream>>>(pe, N, D);

  // GEMM1: sim = relu(cn @ cn^T)           (BT = cn rows)
  k_gemm<0><<<gGemm, tg, 0, stream>>>(cn, (long long)N * D, D,
                                      cn, (long long)N * D, D,
                                      sim, (long long)N * N, N, D,
                                      nullptr, nullptr, nullptr);
  // csum = rowsum(sim)
  k_rowsum<<<B * N, tb, 0, stream>>>(sim, csum, N);

  // GEMM2: vout = posenc @ sim             (sim symmetric -> BT = sim rows)
  k_gemm<1><<<gGemm, tg, 0, stream>>>(pe, 0, D,
                                      sim, (long long)N * N, N,
                                      vout, (long long)N * N, N, N,
                                      nullptr, nullptr, nullptr);

  // GEMM3: counter = softplus(vout @ WexpT^T + csum*w0 + b_exp) -> acat[:, 2048:]
  k_gemm<2><<<gGemm, tg, 0, stream>>>(vout, (long long)N * D, D,
                                      WexpT, 0, D,
                                      acat + D, (long long)N * 2 * D, 2 * D, D,
                                      csum, w0, b_exp);

  // GEMM4: out = acat @ WmergeT^T  (K = 4096, f32 output)
  k_gemm<3><<<gGemm, tg, 0, stream>>>(acat, (long long)N * 2 * D, 2 * D,
                                      WmergeT, 0, 2 * D,
                                      d_out, (long long)N * D, D, 2 * D,
                                      nullptr, nullptr, nullptr);
}
